// MoEGPT2_90572270338152
// MI455X (gfx1250) — matrix operand order repaired and checked by reading, not compiled, not run on hardware
//
#include <hip/hip_runtime.h>

// ---------------------------------------------------------------------------
// MoE GPT2 FFN (top-1 routing), CDNA5 / gfx1250, bf16 WMMA with f32 accum.
// Preprocessing per launch: x -> bf16, w1/w2 -> transposed bf16 (workspace).
// FFN: 64-token tiles, 16 waves; every B fragment feeds 4 WMMAs; x slabs
// staged to LDS via async global->LDS loads (ASYNCcnt) when available.
// ---------------------------------------------------------------------------

typedef __bf16 bf16;
typedef __attribute__((ext_vector_type(16))) __bf16        v16bf;
typedef __attribute__((ext_vector_type(8)))  float         v8f;
typedef __attribute__((ext_vector_type(4)))  unsigned int  v4u;
typedef __attribute__((ext_vector_type(4)))  float         v4f;
typedef int v4i_vs __attribute__((vector_size(16)));   // matches builtin proto

#if __has_builtin(__builtin_amdgcn_global_load_async_to_lds_b128)
#define HAVE_ASYNC_LDS 1
#define GLOBAL_AS1(p) ((v4i_vs __attribute__((address_space(1)))*)(p))
#define LDS_AS3(p)    ((v4i_vs __attribute__((address_space(3)))*)(p))
#else
#define HAVE_ASYNC_LDS 0
#endif

__device__ __forceinline__ void wait_async0() {
#if __has_builtin(__builtin_amdgcn_s_wait_asynccnt)
  __builtin_amdgcn_s_wait_asynccnt(0);
#else
  asm volatile("s_wait_asynccnt 0" ::: "memory");
#endif
}

static constexpr int kT = 16384;   // B*S tokens
static constexpr int kD = 1024;    // model dim
static constexpr int kE = 8;       // experts
static constexpr int kF = 4096;    // expert hidden dim
static constexpr int kMTile = 64;  // token rows per workgroup tile
static constexpr int kCF = 256;    // F-chunk width (16 waves x 16 cols)
static constexpr int kMaxTiles = 264;  // ceil(T/64) + E padding

// Workspace layout: int32 control region, then bf16 tensors.
static constexpr int WS_COUNTS = 0;              // [8]
static constexpr int WS_FILL   = 8;              // [8]
static constexpr int WS_OFFS   = 16;             // [9]
static constexpr int WS_NT     = 25;             // [1]
static constexpr int WS_TEXP   = 64;             // [kMaxTiles]
static constexpr int WS_TSTART = 64 + kMaxTiles; // [kMaxTiles]
static constexpr int WS_IDX    = 8192;           // [kT]
static constexpr int WS_ORDER  = 8192 + kT;      // [kT]
static constexpr size_t WS_XBF_OFF = 262144;                       // 256 KB
static constexpr size_t kXElems  = (size_t)kT * kD;                // 16.7M
static constexpr size_t kWElems  = (size_t)kE * kD * kF;           // 33.5M
static constexpr size_t WS_W1T_OFF = WS_XBF_OFF + kXElems * 2;     // +32 MB
static constexpr size_t WS_W2T_OFF = WS_W1T_OFF + kWElems * 2;     // +64 MB

// ---------------------------------------------------------------------------
// Kernel 1: router  (logits = x @ router_w + router_b ; idx = argmax)
// ---------------------------------------------------------------------------
__global__ void k_router(const float* __restrict__ x,
                         const float* __restrict__ rw,
                         const float* __restrict__ rb,
                         int* __restrict__ ws) {
  int wv = threadIdx.x >> 5;
  int lane = threadIdx.x & 31;
  int t = blockIdx.x * 8 + wv;
  if (t >= kT) return;
  const float* xr = x + (size_t)t * kD;

  float acc[kE];
#pragma unroll
  for (int e = 0; e < kE; ++e) acc[e] = 0.0f;

  for (int i = 0; i < kD / 32; ++i) {
    int d = i * 32 + lane;
    float xv = xr[d];
    const float* wr = rw + (size_t)d * kE;
#pragma unroll
    for (int e = 0; e < kE; ++e) acc[e] += xv * wr[e];
  }
#pragma unroll
  for (int e = 0; e < kE; ++e) {
#pragma unroll
    for (int off = 16; off >= 1; off >>= 1)
      acc[e] += __shfl_xor(acc[e], off, 32);
  }
  if (lane == 0) {
    float best = acc[0] + rb[0];
    int bi = 0;
#pragma unroll
    for (int e = 1; e < kE; ++e) {
      float v = acc[e] + rb[e];
      if (v > best) { best = v; bi = e; }   // strict > == jnp.argmax tie-break
    }
    ws[WS_IDX + t] = bi;
    atomicAdd(&ws[WS_COUNTS + bi], 1);
  }
}

// ---------------------------------------------------------------------------
// Kernel 2: expert offsets, tile descriptors, load-balance loss.
// ---------------------------------------------------------------------------
__global__ void k_offsets(int* __restrict__ ws, float* __restrict__ out_lb) {
  if (threadIdx.x != 0) return;
  int off = 0, nt = 0;
  float lb = 0.0f;
  for (int e = 0; e < kE; ++e) {
    int c = ws[WS_COUNTS + e];
    ws[WS_OFFS + e] = off;
    int ntE = (c + kMTile - 1) / kMTile;
    for (int t2 = 0; t2 < ntE; ++t2) {
      ws[WS_TEXP + nt]   = e;
      ws[WS_TSTART + nt] = off + t2 * kMTile;
      ++nt;
    }
    off += c;
    float u = (float)c / (float)kT;
    float dlt = u - 1.0f / (float)kE;
    lb += dlt * dlt;
  }
  ws[WS_OFFS + kE] = off;
  ws[WS_NT] = nt;
  *out_lb = lb / (float)kE;
}

// ---------------------------------------------------------------------------
// Kernel 3: scatter token ids into per-expert contiguous order[].
// ---------------------------------------------------------------------------
__global__ void k_scatter(int* __restrict__ ws) {
  int t = blockIdx.x * blockDim.x + threadIdx.x;
  if (t >= kT) return;
  int e = ws[WS_IDX + t];
  int pos = atomicAdd(&ws[WS_FILL + e], 1);
  ws[WS_ORDER + ws[WS_OFFS + e] + pos] = t;
}

// ---------------------------------------------------------------------------
// Kernel 4: x fp32 -> bf16 (elementwise, coalesced).
// ---------------------------------------------------------------------------
__global__ void k_cvt_x(const float* __restrict__ x, bf16* __restrict__ xb) {
  size_t i = ((size_t)blockIdx.x * 256 + threadIdx.x) * 4;
  v4f v = *(const v4f*)(x + i);
  bf16* d = xb + i;
  d[0] = (bf16)v.x; d[1] = (bf16)v.y; d[2] = (bf16)v.z; d[3] = (bf16)v.w;
}

// ---------------------------------------------------------------------------
// Kernel 5: tiled transpose + fp32->bf16:  out[e][c][r] = (bf16) in[e][r][c]
// ---------------------------------------------------------------------------
__global__ void k_transpose_cvt(const float* __restrict__ in,
                                bf16* __restrict__ out,
                                int rows, int cols) {
  __shared__ float sh[32][33];
  int tilesC = cols >> 5;
  int perExp = (rows >> 5) * tilesC;
  int bx = blockIdx.x;
  int e  = bx / perExp;
  int rem = bx - e * perExp;
  int r0 = (rem / tilesC) << 5;
  int c0 = (rem % tilesC) << 5;
  const float* ine = in + (size_t)e * rows * cols;
  bf16* oute = out + (size_t)e * rows * cols;

  int tid = threadIdx.x;
#pragma unroll
  for (int it = 0; it < 4; ++it) {
    int idx = it * 256 + tid;
    int r = idx >> 5, c = idx & 31;
    sh[r][c] = ine[(size_t)(r0 + r) * cols + c0 + c];
  }
  __syncthreads();
#pragma unroll
  for (int it = 0; it < 4; ++it) {
    int idx = it * 256 + tid;
    int r = idx >> 5, c = idx & 31;
    oute[(size_t)(c0 + r) * rows + r0 + c] = (bf16)sh[c][r];
  }
}

// ---------------------------------------------------------------------------
// WMMA fragment helpers.
// 16-bit A 16x32: lane L holds row m=L%16, hi=L/16;
// element e <-> K=hi*16+e (matches B order; two 16B runs).
// B 32x16: lane holds column n=L%16; element e <-> K=hi*16+e (contiguous in
// the transposed bf16 weight layout -> two 16B loads).
// ---------------------------------------------------------------------------
__device__ __forceinline__ v16bf load_a_frag(const bf16* rowPtr, int kb, int hi) {
  union { v16bf v; v4u q[2]; } u;
  u.q[0] = *(const v4u*)(rowPtr + kb + hi * 16);
  u.q[1] = *(const v4u*)(rowPtr + kb + hi * 16 + 8);
  return u.v;
}

__device__ __forceinline__ v16bf load_b_frag16(const bf16* p) {
  union { v16bf v; v4u q[2]; } u;
  u.q[0] = *(const v4u*)(p);
  u.q[1] = *(const v4u*)(p + 8);
  return u.v;
}

// ---------------------------------------------------------------------------
// Kernel 6: fused grouped FFN.
//   h = relu(x_tile @ w1[e] + b1[e]) ; y = h @ w2[e] + b2[e]
// Per workgroup: 64 tokens x all 1024 output columns, 16 waves (512 thr).
// F in 256-wide chunks; x staged per 128-K slab (async->LDS when available);
// h lives in LDS as bf16. Each B fragment feeds 4 WMMAs (4 M-subtiles).
// ---------------------------------------------------------------------------
__launch_bounds__(512, 1)
__global__ void k_moe_ffn(const bf16* __restrict__ xbf,   // [T][D] bf16
                          const bf16* __restrict__ w1T,   // [E][F][D] bf16
                          const float* __restrict__ b1,
                          const bf16* __restrict__ w2T,   // [E][D][F] bf16
                          const float* __restrict__ b2,
                          const int* __restrict__ ws,
                          float* __restrict__ out) {
  __shared__ bf16 sh_xk[kMTile][128 + 8];   // 17.0 KB  x K-slab
  __shared__ bf16 sh_h [kMTile][kCF + 8];   // 33.0 KB  h chunk
  __shared__ int  sh_tok[kMTile];           // token ids (-1 = pad)
  __shared__ int  sh_roff[kMTile];          // clamped token*kD
  __shared__ int  sh_meta[2];               // [expert, nvalid]

  int tile = blockIdx.x;
  if (tile >= ws[WS_NT]) return;

  int tid  = threadIdx.x;
  int wv   = tid >> 5;         // 0..15
  int lane = tid & 31;
  int n    = lane & 15;        // A-row index AND B/C column index
  int hi   = lane >> 4;

  if (tid < kMTile) {
    int e     = ws[WS_TEXP + tile];
    int start = ws[WS_TSTART + tile];
    int end   = ws[WS_OFFS + e + 1];
    if (tid == 0) { sh_meta[0] = e; sh_meta[1] = min(kMTile, end - start); }
    int tok = (start + tid < end) ? ws[WS_ORDER + start + tid] : -1;
    sh_tok[tid]  = tok;
    sh_roff[tid] = (tok < 0 ? 0 : tok) * kD;
  }
  __syncthreads();

  int e = sh_meta[0];
  int nvalid = sh_meta[1];
  const bf16*  w1e = w1T + (size_t)e * kF * kD;  // [F][D]
  const bf16*  w2e = w2T + (size_t)e * kD * kF;  // [D][F]
  const float* b1e = b1 + (size_t)e * kF;
  const float* b2e = b2 + (size_t)e * kD;

  // Loop-invariant staging geometry: this thread copies two 16B segments of
  // the 64x128 x slab (1024 segments total / 512 threads).
  int sr0 = tid >> 4,          sk0 = (tid & 15) << 3;
  int sr1 = (tid + 512) >> 4,  sk1 = (tid & 15) << 3;   // (tid+512)&15 == tid&15
  const bf16* gsrc0 = xbf + (size_t)sh_roff[sr0] + sk0;
  const bf16* gsrc1 = xbf + (size_t)sh_roff[sr1] + sk1;
  bf16* ldst0 = &sh_xk[sr0][sk0];
  bf16* ldst1 = &sh_xk[sr1][sk1];

  // Loop-invariant A-row LDS base pointers per M-subtile.
  const bf16* aRow[4];
  const bf16* hRow[4];
#pragma unroll
  for (int ms = 0; ms < 4; ++ms) {
    aRow[ms] = &sh_xk[ms * 16 + n][0];
    hRow[ms] = &sh_h [ms * 16 + n][0];
  }

  v8f yacc[4][4];
#pragma unroll
  for (int ms = 0; ms < 4; ++ms)
#pragma unroll
    for (int nt = 0; nt < 4; ++nt)
      yacc[ms][nt] = (v8f){0, 0, 0, 0, 0, 0, 0, 0};

  int cb = wv * 64;            // this wave's output column base (4 subtiles)

  for (int f0 = 0; f0 < kF; f0 += kCF) {
    // ---- GEMM1: wave computes h[:, fb..fb+16) for 4 M-subtiles ----------
    int fb = f0 + wv * 16;
    const bf16* w1row = w1e + (size_t)(fb + n) * kD + hi * 16;
    v8f hacc[4];
#pragma unroll
    for (int ms = 0; ms < 4; ++ms) hacc[ms] = (v8f){0, 0, 0, 0, 0, 0, 0, 0};

    for (int kb0 = 0; kb0 < kD; kb0 += 128) {
      __syncthreads();  // previous slab consumers / previous chunk GEMM2 done
#if HAVE_ASYNC_LDS
      // Async global->LDS copy (ASYNCcnt), no VGPR round-trip.
      __builtin_amdgcn_global_load_async_to_lds_b128(
          GLOBAL_AS1(gsrc0 + kb0), LDS_AS3(ldst0), 0, 0);
      __builtin_amdgcn_global_load_async_to_lds_b128(
          GLOBAL_AS1(gsrc1 + kb0), LDS_AS3(ldst1), 0, 0);
      wait_async0();
#else
      *(v4u*)ldst0 = *(const v4u*)(gsrc0 + kb0);
      *(v4u*)ldst1 = *(const v4u*)(gsrc1 + kb0);
#endif
      __syncthreads();
#pragma unroll
      for (int kl = 0; kl < 128; kl += 32) {
        v16bf a[4];
#pragma unroll
        for (int ms = 0; ms < 4; ++ms)
          a[ms] = load_a_frag(aRow[ms], kl, hi);
        v16bf bw = load_b_frag16(w1row + kb0 + kl);
#pragma unroll
        for (int ms = 0; ms < 4; ++ms)
          hacc[ms] = __builtin_amdgcn_wmma_f32_16x16x32_bf16(
              false, a[ms], false, bw, (short)0, hacc[ms], false, false);
      }
    }

    // relu + bias -> sh_h (C layout: element g -> row g + hi*8, col n)
    float bias1 = b1e[fb + n];
#pragma unroll
    for (int ms = 0; ms < 4; ++ms)
#pragma unroll
      for (int g = 0; g < 8; ++g) {
        int row = ms * 16 + g + hi * 8;
        float hv = fmaxf(hacc[ms][g] + bias1, 0.0f);
        sh_h[row][wv * 16 + n] = (bf16)hv;
      }
    __syncthreads();  // h chunk visible to all waves

    // ---- GEMM2: yacc += h_chunk @ w2[f0:f0+kCF, cb:cb+64) ---------------
#pragma unroll 2
    for (int kf = 0; kf < kCF; kf += 32) {
      v16bf a2[4];
#pragma unroll
      for (int ms = 0; ms < 4; ++ms)
        a2[ms] = load_a_frag(hRow[ms], kf, hi);
#pragma unroll
      for (int nt = 0; nt < 4; ++nt) {
        v16bf bw = load_b_frag16(
            w2e + (size_t)(cb + nt * 16 + n) * kF + f0 + kf + hi * 16);
#pragma unroll
        for (int ms = 0; ms < 4; ++ms)
          yacc[ms][nt] = __builtin_amdgcn_wmma_f32_16x16x32_bf16(
              false, a2[ms], false, bw, (short)0, yacc[ms][nt], false, false);
      }
    }
  }

  // Epilogue: y + b2 -> out, scattered by token id. Coalesced across lanes.
#pragma unroll
  for (int nt = 0; nt < 4; ++nt) {
    int col = cb + nt * 16 + n;
    float bias2 = b2e[col];
#pragma unroll
    for (int ms = 0; ms < 4; ++ms)
#pragma unroll
      for (int g = 0; g < 8; ++g) {
        int row = ms * 16 + g + hi * 8;
        if (row < nvalid) {
          int tok = sh_tok[row];
          out[(size_t)tok * kD + col] = yacc[ms][nt][g] + bias2;
        }
      }
  }
}

// ---------------------------------------------------------------------------
// Launch
// ---------------------------------------------------------------------------
extern "C" void kernel_launch(void* const* d_in, const int* in_sizes, int n_in,
                              void* d_out, int out_size, void* d_ws, size_t ws_size,
                              hipStream_t stream) {
  const float* x  = (const float*)d_in[0];
  const float* rw = (const float*)d_in[1];
  const float* rb = (const float*)d_in[2];
  const float* w1 = (const float*)d_in[3];
  const float* b1 = (const float*)d_in[4];
  const float* w2 = (const float*)d_in[5];
  const float* b2 = (const float*)d_in[6];
  float* out = (float*)d_out;
  int* ws = (int*)d_ws;

  bf16* xbf = (bf16*)((char*)d_ws + WS_XBF_OFF);   // [T][D]
  bf16* w1T = (bf16*)((char*)d_ws + WS_W1T_OFF);   // [E][F][D]
  bf16* w2T = (bf16*)((char*)d_ws + WS_W2T_OFF);   // [E][D][F]

  // zero counts/fill/offs/numTiles (graph-capture legal)
  (void)hipMemsetAsync(ws, 0, 64 * sizeof(int), stream);

  // Preprocess: x -> bf16; weights -> transposed bf16.
  k_cvt_x<<<(int)(kXElems / 1024), 256, 0, stream>>>(x, xbf);
  int blocksW = kE * (kD / 32) * (kF / 32);   // 32768
  k_transpose_cvt<<<blocksW, 256, 0, stream>>>(w1, w1T, kD, kF);
  k_transpose_cvt<<<blocksW, 256, 0, stream>>>(w2, w2T, kF, kD);

  k_router <<<kT / 8, 256, 0, stream>>>(x, rw, rb, ws);
  k_offsets<<<1, 32, 0, stream>>>(ws, out + (size_t)kT * kD);  // lb_loss slot
  k_scatter<<<kT / 256, 256, 0, stream>>>(ws);
  k_moe_ffn<<<kMaxTiles, 512, 0, stream>>>(xbf, w1T, b1, w2T, b2, ws, out);
}
